// C_loss_69415261438022
// MI455X (gfx1250) — compile-verified, matrix operand order checked by reading, not run on hardware
//
#include <hip/hip_runtime.h>

typedef _Float16 v16h __attribute__((ext_vector_type(16)));
typedef _Float16 v8h  __attribute__((ext_vector_type(8)));
typedef _Float16 v4h  __attribute__((ext_vector_type(4)));
typedef float    v8f  __attribute__((ext_vector_type(8)));
typedef float    v4f  __attribute__((ext_vector_type(4)));

#define N_ROWS 16384
#define DIM    128
#define BLK_M  128   // rows of A per block
#define BLK_N  64    // rows of B (cols of sim) per block
// exp(-sim/T) = exp(-2*sim) = exp2(sim * (-2*log2(e)))
#define NEG2_LOG2E (-2.885390081777926815f)

// ---------------------------------------------------------------------------
// Kernel 1: L2-normalize rows of both inputs, convert to f16.
// One wave32 per row: lane l holds elements 4l..4l+3 (float4).
// Also zeroes the fp64 accumulator (block 0, thread 0).
// ---------------------------------------------------------------------------
__global__ __launch_bounds__(256) void normalize_f16_kernel(
    const float* __restrict__ selfP, const float* __restrict__ posP,
    _Float16* __restrict__ aH, _Float16* __restrict__ bH,
    double* __restrict__ accum)
{
    if (blockIdx.x == 0 && threadIdx.x == 0) *accum = 0.0;

    int wave = threadIdx.x >> 5;
    int lane = threadIdx.x & 31;
    int row  = blockIdx.x * 8 + wave;          // 0 .. 2N-1

    const float* src; _Float16* dst; int r;
    if (row < N_ROWS) { src = selfP; dst = aH; r = row; }
    else              { src = posP;  dst = bH; r = row - N_ROWS; }

    v4f v = *(const v4f*)(src + (size_t)r * DIM + lane * 4);
    float ss = v.x * v.x + v.y * v.y + v.z * v.z + v.w * v.w;
    #pragma unroll
    for (int off = 16; off >= 1; off >>= 1)
        ss += __shfl_xor(ss, off, 32);

    float inv = 1.0f / fmaxf(sqrtf(ss), 1e-12f);
    v4h h;
    h.x = (_Float16)(v.x * inv);
    h.y = (_Float16)(v.y * inv);
    h.z = (_Float16)(v.z * inv);
    h.w = (_Float16)(v.w * inv);
    *(v4h*)(dst + (size_t)r * DIM + lane * 4) = h;
}

// ---------------------------------------------------------------------------
// Helper: load one 16x32 f16 WMMA operand fragment from a row-major LDS tile.
// Per lane: two contiguous b128 LDS reads (K 0-7/16-23 for lanes 0-15,
// +8 halves for lanes 16-31) -> matches the CDNA5 16-bit A/B VGPR layout.
// ---------------------------------------------------------------------------
__device__ __forceinline__ v16h load_frag(const _Float16* base, int kc)
{
    v8h lo = *(const v8h*)(base + kc * 32);
    v8h hi = *(const v8h*)(base + kc * 32 + 16);
    return __builtin_shufflevector(lo, hi,
        0,1,2,3,4,5,6,7,8,9,10,11,12,13,14,15);
}

// ---------------------------------------------------------------------------
// Kernel 2: fused WMMA GEMM (sim = a . b^T) + exp(-2*sim) * (labels differ)
// reduced to a single fp64 accumulator. sim is never materialized.
// Block: 256 threads (8 wave32), computes a 128x64 block of sim.
// Each wave owns one 16-row strip; column tiles are processed in pairs with
// two independent accumulator chains so WMMA issue is not serialized on LDS.
// ---------------------------------------------------------------------------
__global__ __launch_bounds__(256) void gemm_exp_reduce_kernel(
    const _Float16* __restrict__ aH, const _Float16* __restrict__ bH,
    const int* __restrict__ labels, double* __restrict__ accum)
{
    __shared__ _Float16 ldsA[BLK_M * DIM];   // 32 KB
    __shared__ _Float16 ldsB[BLK_N * DIM];   // 16 KB
    __shared__ int rowLab[BLK_M];
    __shared__ int colLab[BLK_N];
    __shared__ float waveSum[8];

    const int tid     = threadIdx.x;
    const int rowBase = blockIdx.y * BLK_M;
    const int colBase = blockIdx.x * BLK_N;

    // --- stage f16 tiles into LDS (rows are contiguous, b128 chunks) ---
    {
        const uint4* aG = (const uint4*)(aH + (size_t)rowBase * DIM);
        const uint4* bG = (const uint4*)(bH + (size_t)colBase * DIM);
        uint4* lA = (uint4*)ldsA;
        uint4* lB = (uint4*)ldsB;
        #pragma unroll
        for (int i = tid; i < BLK_M * DIM / 8; i += 256) lA[i] = aG[i];
        #pragma unroll
        for (int i = tid; i < BLK_N * DIM / 8; i += 256) lB[i] = bG[i];
        if (tid < BLK_M)                 rowLab[tid]         = labels[rowBase + tid];
        else if (tid < BLK_M + BLK_N)    colLab[tid - BLK_M] = labels[colBase + tid - BLK_M];
    }
    __syncthreads();

    const int wave   = tid >> 5;
    const int lane   = tid & 31;
    const int laneHi = lane >> 4;   // 0: K-low half, 1: K-high half
    const int lane15 = lane & 15;

    // --- load this wave's 4 A fragments (16x32 f16 each), reused 4x ---
    v16h afrag[4];
    {
        const _Float16* base = ldsA + (wave * 16 + lane15) * DIM + laneHi * 8;
        #pragma unroll
        for (int kc = 0; kc < 4; ++kc) afrag[kc] = load_frag(base, kc);
    }

    // C/D 16x16 f32 layout: VGPR v, lanes<16 -> M=v, lanes>=16 -> M=v+8
    int myRowLab[8];
    #pragma unroll
    for (int v = 0; v < 8; ++v)
        myRowLab[v] = rowLab[wave * 16 + laneHi * 8 + v];

    float lsum = 0.0f;

    // --- column tiles in pairs: two independent accumulator chains ---
    #pragma unroll
    for (int cp = 0; cp < 2; ++cp) {
        const int col0 = (2 * cp + 0) * 16 + lane15;
        const int col1 = (2 * cp + 1) * 16 + lane15;
        const _Float16* bb0 = ldsB + col0 * DIM + laneHi * 8;
        const _Float16* bb1 = ldsB + col1 * DIM + laneHi * 8;

        // preload all B fragments for both tiles (16 x ds_load_b128)
        v16h bf0[4], bf1[4];
        #pragma unroll
        for (int kc = 0; kc < 4; ++kc) {
            bf0[kc] = load_frag(bb0, kc);
            bf1[kc] = load_frag(bb1, kc);
        }

        v8f acc0 = {};
        v8f acc1 = {};
        #pragma unroll
        for (int kc = 0; kc < 4; ++kc) {
            acc0 = __builtin_amdgcn_wmma_f32_16x16x32_f16(
                false, afrag[kc], false, bf0[kc], (short)0, acc0, false, false);
            acc1 = __builtin_amdgcn_wmma_f32_16x16x32_f16(
                false, afrag[kc], false, bf1[kc], (short)0, acc1, false, false);
        }

        const int cl0 = colLab[col0];
        const int cl1 = colLab[col1];
        #pragma unroll
        for (int v = 0; v < 8; ++v) {
            float e0 = __builtin_amdgcn_exp2f(NEG2_LOG2E * acc0[v]);
            float e1 = __builtin_amdgcn_exp2f(NEG2_LOG2E * acc1[v]);
            lsum += (myRowLab[v] != cl0) ? e0 : 0.0f;
            lsum += (myRowLab[v] != cl1) ? e1 : 0.0f;
        }
    }

    // --- hierarchical reduction: wave32 -> block -> fp64 global atomic ---
    #pragma unroll
    for (int off = 16; off >= 1; off >>= 1)
        lsum += __shfl_xor(lsum, off, 32);
    if (lane == 0) waveSum[wave] = lsum;
    __syncthreads();
    if (tid == 0) {
        double s = 0.0;
        #pragma unroll
        for (int i = 0; i < 8; ++i) s += (double)waveSum[i];
        atomicAdd(accum, s);
    }
}

// ---------------------------------------------------------------------------
// Kernel 3: finalize — divide by N*(N-1), emit fp32 scalar.
// ---------------------------------------------------------------------------
__global__ void finalize_kernel(const double* __restrict__ accum,
                                float* __restrict__ out)
{
    if (threadIdx.x == 0 && blockIdx.x == 0) {
        double denom = (double)N_ROWS * (double)(N_ROWS - 1);
        out[0] = (float)(accum[0] / denom);
    }
}

// ---------------------------------------------------------------------------
extern "C" void kernel_launch(void* const* d_in, const int* in_sizes, int n_in,
                              void* d_out, int out_size, void* d_ws, size_t ws_size,
                              hipStream_t stream)
{
    const float* selfP = (const float*)d_in[0];
    const float* posP  = (const float*)d_in[1];
    const int*   lab   = (const int*)d_in[2];
    float* out = (float*)d_out;

    char* ws = (char*)d_ws;
    _Float16* aH    = (_Float16*)ws;                                   // 4 MB
    _Float16* bH    = (_Float16*)(ws + (size_t)N_ROWS * DIM * 2);      // 4 MB
    double*   accum = (double*)  (ws + (size_t)2 * N_ROWS * DIM * 2);  // 8 B

    normalize_f16_kernel<<<(2 * N_ROWS) / 8, 256, 0, stream>>>(
        selfP, posP, aH, bH, accum);

    dim3 grid(N_ROWS / BLK_N, N_ROWS / BLK_M);   // 256 x 128 blocks
    gemm_exp_reduce_kernel<<<grid, 256, 0, stream>>>(aH, bH, lab, accum);

    finalize_kernel<<<1, 32, 0, stream>>>(accum, out);
}
